// LSTM_81398220194434
// MI455X (gfx1250) — matrix operand order repaired and checked by reading, not compiled, not run on hardware
//
#include <hip/hip_runtime.h>
#include <hip/hip_bf16.h>

#define T_STEPS 2048
#define BATCH   32
#define HID     512
#define NWG     32
#define NTHREADS 512
// LDS: wih 64KB + x 32KB + gates(2 K-halves) 16KB = 112KB (of 320KB/WGP)
#define LDS_WIH_OFF 0
#define LDS_X_OFF   (64 * 512 * 2)
#define LDS_G_OFF   (LDS_X_OFF + 32 * 512 * 2)
#define SMEM_BYTES  (LDS_G_OFF + 2 * 4 * 32 * 16 * 4)

typedef __attribute__((ext_vector_type(16))) __bf16 v16bf;
typedef __attribute__((ext_vector_type(8)))  float  v8f;

union ABFrag { v16bf v; uint4 q[2]; };

__device__ __forceinline__ unsigned short f32_to_bf16_rn(float f) {
  unsigned u = __float_as_uint(f);
  u += 0x7FFFu + ((u >> 16) & 1u);
  return (unsigned short)(u >> 16);
}

__device__ __forceinline__ float fast_sigmoid(float x) {
  return __builtin_amdgcn_rcpf(1.f + __expf(-x));
}
__device__ __forceinline__ float fast_tanh(float x) {
  return 1.f - 2.f * __builtin_amdgcn_rcpf(__expf(2.f * x) + 1.f);
}

// CDNA5 async copy: GVS mode  mem = SADDR(u64) + VADDR(u32); LDS dest per lane.
__device__ __forceinline__ void async_ld_b128(unsigned lds_addr, unsigned voff,
                                              const void* sbase) {
  asm volatile("global_load_async_to_lds_b128 %0, %1, %2"
               :: "v"(lds_addr), "v"(voff), "s"(sbase) : "memory");
}
__device__ __forceinline__ void wait_async0() {
  asm volatile("s_wait_asynccnt 0" ::: "memory");
}

__global__ void cvt_f32_to_bf16(const float* __restrict__ src,
                                unsigned short* __restrict__ dst, int n) {
  for (int i = blockIdx.x * blockDim.x + threadIdx.x; i < n;
       i += gridDim.x * blockDim.x)
    dst[i] = f32_to_bf16_rn(src[i]);
}

__global__ void __launch_bounds__(NTHREADS)
lstm_layer_persistent(const unsigned short* __restrict__ xin,   // [T,B,512] bf16
                      const unsigned short* __restrict__ Wih,   // [2048,512] bf16
                      const unsigned short* __restrict__ Whh,   // [2048,512] bf16
                      unsigned short* __restrict__ hring,       // [2,B,512] bf16
                      unsigned short* __restrict__ ybf16,       // layer0 out (or null)
                      float* __restrict__ yf32,                 // layer1 out (or null)
                      float* __restrict__ hy,
                      float* __restrict__ cy,
                      unsigned* __restrict__ bar) {
  extern __shared__ unsigned char smem[];
  unsigned short* lds_wih = (unsigned short*)(smem + LDS_WIH_OFF); // [64][512]
  unsigned short* lds_x   = (unsigned short*)(smem + LDS_X_OFF);   // [32][512]
  float*          lds_g   = (float*)(smem + LDS_G_OFF);            // [2][4][32][16]

  const int tid  = threadIdx.x;
  const int n0   = blockIdx.x * 16;    // 16 hidden columns per WG
  const int wid  = tid >> 5;           // 16 waves (wave32)
  const int lane = tid & 31;
  const int gate = wid >> 2;           // 0..3 (i,f,g,o)
  const int tm   = (wid >> 1) & 1;     // batch rows 0-15 / 16-31
  const int kh   = wid & 1;            // K half: 0 -> K 0..255, 1 -> K 256..511
  const int hi   = (lane < 16) ? 0 : 1;
  const int l15  = lane & 15;

  const unsigned ldsx_base = (unsigned)(uintptr_t)(void*)lds_x;

  // ---- Preload W_ih slice (64 rows x 512 K) into LDS.
  for (int idx = tid; idx < 64 * 64; idx += NTHREADS) {
    int row  = idx >> 6;                             // gate*16 + n_local
    int grow = (row >> 4) * HID + n0 + (row & 15);   // global weight row
    ((uint4*)lds_wih)[idx] = ((const uint4*)Wih)[grow * 64 + (idx & 63)];
  }

  // ---- Preload this wave's K-half of W_hh fragments into registers.
  // B (32x16): lane = N col (l15); K chunks {0-7,16-23} (lo lanes) / {8-15,24-31}.
  ABFrag bh[8];
  {
    const uint4* wp = (const uint4*)Whh +
                      (size_t)(gate * HID + n0 + l15) * 64 + kh * 32;
#pragma unroll
    for (int kk = 0; kk < 8; ++kk) {
      bh[kk].q[0] = wp[kk * 4 + hi];
      bh[kk].q[1] = wp[kk * 4 + hi + 2];
    }
  }

  // Per-lane fragment pointers (K-half baked in).
  // A (16x32): lane = M row; K chunks {0-7,16-23} (lo lanes) / {8-15,24-31}.
  const int arow = l15 + tm * 16;
  const uint4* axp = (const uint4*)(lds_x + arow * HID) + kh * 32;
  const uint4* bip = (const uint4*)(lds_wih + (gate * 16 + l15) * HID) + kh * 32;
  const int aoff = hi;        // +8 elems  = 1 uint4
  const int boff = hi;        // +8 elems  = 1 uint4

  // x-part partial GEMM over this wave's K half (dual accumulator chains).
  auto xpart = [&]() -> v8f {
    v8f s1 = {}, s2 = {};
#pragma unroll
    for (int kk = 0; kk < 8; kk += 2) {
      ABFrag a1, b1, a2, b2;
      a1.q[0] = axp[kk * 4 + aoff];       a1.q[1] = axp[kk * 4 + aoff + 2];
      b1.q[0] = bip[kk * 4 + boff];       b1.q[1] = bip[kk * 4 + boff + 2];
      a2.q[0] = axp[(kk + 1) * 4 + aoff]; a2.q[1] = axp[(kk + 1) * 4 + aoff + 2];
      b2.q[0] = bip[(kk + 1) * 4 + boff]; b2.q[1] = bip[(kk + 1) * 4 + boff + 2];
      s1 = __builtin_amdgcn_wmma_f32_16x16x32_bf16(false, a1.v, false, b1.v,
                                                   (short)0, s1, false, false);
      s2 = __builtin_amdgcn_wmma_f32_16x16x32_bf16(false, a2.v, false, b2.v,
                                                   (short)0, s2, false, false);
    }
    return s1 + s2;
  };

  // ---- Stage x(0), then its partial gate contribution.
  for (int idx = tid; idx < BATCH * HID / 8; idx += NTHREADS)
    async_ld_b128(ldsx_base + idx * 16, idx * 16, xin);
  wait_async0();
  __syncthreads();                                   // covers wih preload too
  v8f accx = xpart();

  float c0 = 0.f, c1 = 0.f;                          // cell state (tid<256 only)
  const int em = tid >> 3;                           // elementwise row
  const int en = (tid << 1) & 15;                    // elementwise col pair

  for (int t = 0; t < T_STEPS; ++t) {
    // ---- Wait until every WG published h(t-1) (epoch barrier, acquire).
    if (tid == 0) {
      unsigned target = (unsigned)NWG * (unsigned)t;
      while (__hip_atomic_load(bar, __ATOMIC_ACQUIRE,
                               __HIP_MEMORY_SCOPE_AGENT) < target)
        __builtin_amdgcn_s_sleep(1);
    }
    __syncthreads();

    // ---- Kick off x(t+1) staging; DMA overlaps the whole h-part + cell math.
    if (t + 1 < T_STEPS) {
      const unsigned short* gx = xin + (size_t)(t + 1) * BATCH * HID;
      for (int idx = tid; idx < BATCH * HID / 8; idx += NTHREADS)
        async_ld_b128(ldsx_base + idx * 16, idx * 16, gx);
    }

    // ---- h-part GEMM, critical path: A direct from L2, B from registers.
    const uint4* hp = (const uint4*)(hring + (size_t)(t & 1) * BATCH * HID) +
                      (size_t)arow * 64 + kh * 32;
    v8f s1 = accx, s2 = {};
#pragma unroll
    for (int kk = 0; kk < 8; kk += 2) {
      ABFrag a1, a2;
      a1.q[0] = hp[kk * 4 + aoff];       a1.q[1] = hp[kk * 4 + aoff + 2];
      a2.q[0] = hp[(kk + 1) * 4 + aoff]; a2.q[1] = hp[(kk + 1) * 4 + aoff + 2];
      s1 = __builtin_amdgcn_wmma_f32_16x16x32_bf16(false, a1.v, false, bh[kk].v,
                                                   (short)0, s1, false, false);
      s2 = __builtin_amdgcn_wmma_f32_16x16x32_bf16(false, a2.v, false,
                                                   bh[kk + 1].v, (short)0, s2,
                                                   false, false);
    }
    v8f g = s1 + s2;

    // Scatter partial gate tile: VGPR r -> M = 16*tm + 8*hi + r, N = l15.
    float* gbuf = lds_g + kh * (4 * BATCH * 16);
#pragma unroll
    for (int r = 0; r < 8; ++r)
      gbuf[(gate * BATCH + (tm * 16 + hi * 8 + r)) * 16 + l15] = g[r];
    __syncthreads();

    // ---- Elementwise LSTM cell: first 8 waves, 2 adjacent columns/thread.
    if (tid < 256) {
      float* g0 = lds_g;
      float* g1 = lds_g + 4 * BATCH * 16;
      int base = em * 16 + en;
      float2 ig = *(float2*)&g0[0 * BATCH * 16 + base];
      float2 fg = *(float2*)&g0[1 * BATCH * 16 + base];
      float2 gg = *(float2*)&g0[2 * BATCH * 16 + base];
      float2 og = *(float2*)&g0[3 * BATCH * 16 + base];
      float2 i1 = *(float2*)&g1[0 * BATCH * 16 + base];
      float2 f1 = *(float2*)&g1[1 * BATCH * 16 + base];
      float2 q1 = *(float2*)&g1[2 * BATCH * 16 + base];
      float2 o1 = *(float2*)&g1[3 * BATCH * 16 + base];
      ig.x += i1.x; ig.y += i1.y; fg.x += f1.x; fg.y += f1.y;
      gg.x += q1.x; gg.y += q1.y; og.x += o1.x; og.y += o1.y;
      float ca = fast_sigmoid(fg.x) * c0 + fast_sigmoid(ig.x) * fast_tanh(gg.x);
      float cb = fast_sigmoid(fg.y) * c1 + fast_sigmoid(ig.y) * fast_tanh(gg.y);
      float ha = fast_sigmoid(og.x) * fast_tanh(ca);
      float hb = fast_sigmoid(og.y) * fast_tanh(cb);
      c0 = ca; c1 = cb;
      int col = n0 + en;
      size_t oidx = (size_t)t * BATCH * HID + (size_t)em * HID + col;
      unsigned hpk = (unsigned)f32_to_bf16_rn(ha) |
                     ((unsigned)f32_to_bf16_rn(hb) << 16);
      *(unsigned*)&hring[(size_t)((t + 1) & 1) * BATCH * HID + em * HID + col] = hpk;
      if (ybf16) *(unsigned*)&ybf16[oidx] = hpk;
      if (yf32)  *(float2*)&yf32[oidx] = make_float2(ha, hb);
      if (t == T_STEPS - 1) {
        *(float2*)&hy[em * HID + col] = make_float2(ha, hb);
        *(float2*)&cy[em * HID + col] = make_float2(ca, cb);
      }
    }

    // ---- Arrive (release h(t)), then finish the overlapped x(t+1) partials.
    __threadfence();
    __syncthreads();
    if (tid == 0)
      __hip_atomic_fetch_add(bar, 1u, __ATOMIC_RELEASE,
                             __HIP_MEMORY_SCOPE_AGENT);
    if (t + 1 < T_STEPS) {
      wait_async0();
      __syncthreads();
      accx = xpart();
    }
  }
}

extern "C" void kernel_launch(void* const* d_in, const int* in_sizes, int n_in,
                              void* d_out, int out_size, void* d_ws, size_t ws_size,
                              hipStream_t stream) {
  (void)in_sizes; (void)n_in; (void)out_size; (void)ws_size;
  const float* x    = (const float*)d_in[0];
  const float* Wih0 = (const float*)d_in[1];
  const float* Whh0 = (const float*)d_in[2];
  const float* Wih1 = (const float*)d_in[3];
  const float* Whh1 = (const float*)d_in[4];

  const size_t WELEMS = (size_t)4 * HID * HID;       // 2048*512
  const size_t XELEMS = (size_t)T_STEPS * BATCH * HID;

  unsigned short* ws    = (unsigned short*)d_ws;
  unsigned short* wih0b = ws;
  unsigned short* whh0b = wih0b + WELEMS;
  unsigned short* wih1b = whh0b + WELEMS;
  unsigned short* whh1b = wih1b + WELEMS;
  unsigned short* xb    = whh1b + WELEMS;
  unsigned short* y0b   = xb + XELEMS;
  unsigned short* hring = y0b + XELEMS;
  unsigned*       bar   = (unsigned*)(hring + 2 * BATCH * HID);

  float* out = (float*)d_out;
  float* y1  = out;                       // [T,B,H]
  float* hy  = out + XELEMS;              // [2,B,H]
  float* cy  = hy + 2 * BATCH * HID;      // [2,B,H]

  (void)hipFuncSetAttribute((const void*)lstm_layer_persistent,
                            hipFuncAttributeMaxDynamicSharedMemorySize,
                            SMEM_BYTES);

  cvt_f32_to_bf16<<<512, 256, 0, stream>>>(Wih0, wih0b, (int)WELEMS);
  cvt_f32_to_bf16<<<512, 256, 0, stream>>>(Whh0, whh0b, (int)WELEMS);
  cvt_f32_to_bf16<<<512, 256, 0, stream>>>(Wih1, wih1b, (int)WELEMS);
  cvt_f32_to_bf16<<<512, 256, 0, stream>>>(Whh1, whh1b, (int)WELEMS);
  cvt_f32_to_bf16<<<4096, 256, 0, stream>>>(x, xb, (int)XELEMS);

  // Layer 0
  hipMemsetAsync(hring, 0, 2 * BATCH * HID * sizeof(unsigned short), stream);
  hipMemsetAsync(bar, 0, sizeof(unsigned), stream);
  lstm_layer_persistent<<<NWG, NTHREADS, SMEM_BYTES, stream>>>(
      xb, wih0b, whh0b, hring, y0b, nullptr, hy, cy, bar);

  // Layer 1
  hipMemsetAsync(hring, 0, 2 * BATCH * HID * sizeof(unsigned short), stream);
  hipMemsetAsync(bar, 0, sizeof(unsigned), stream);
  lstm_layer_persistent<<<NWG, NTHREADS, SMEM_BYTES, stream>>>(
      y0b, wih1b, whh1b, hring, nullptr, y1,
      hy + BATCH * HID, cy + BATCH * HID, bar);
}
